// Shared_82119774699743
// MI455X (gfx1250) — matrix operand order repaired and checked by reading, not compiled, not run on hardware
//
#include <hip/hip_runtime.h>
#include <hip/hip_bf16.h>

typedef _Float16 f16;
typedef __attribute__((ext_vector_type(16))) _Float16 v16h;
typedef __attribute__((ext_vector_type(8)))  float    v8f;

#define BN_EPS 1e-5f

// A-fragment element->K map for 16-bit 16x32 A (contiguous, matches B):
// lanes 0-15: e -> K=e ; lanes 16-31: K=16+e
__device__ __forceinline__ int a_k(int lane, int e) {
  return e + ((lane >= 16) ? 16 : 0);
}

__device__ __forceinline__ v16h ldfragA(const f16* __restrict__ buf, int frag, int lane) {
  return *(const v16h*)(buf + ((size_t)frag * 32 + lane) * 16);
}

__device__ __forceinline__ float elu(float v) {
  return v > 0.f ? v : (__expf(v) - 1.f);
}

__device__ __forceinline__ v8f wmma_f16(v16h a, v16h b, v8f c) {
  // D = A(16x32 f16) x B(32x16 f16) + C(16x16 f32)
  return __builtin_amdgcn_wmma_f32_16x16x32_f16(false, a, false, b, (short)0, c, false, false);
}

// ---------------------------------------------------------------------------
// Prep: fold BN scale into conv weights, build pre-swizzled f16 A-fragments,
// compute per-channel betas, convert Wf to f16.
// ---------------------------------------------------------------------------
__global__ void prep_kernel(
    const float* __restrict__ Wt, const float* __restrict__ bt,
    const float* __restrict__ g1, const float* __restrict__ b1n,
    const float* __restrict__ m1, const float* __restrict__ v1,
    const float* __restrict__ Ws, const float* __restrict__ bs,
    const float* __restrict__ g2, const float* __restrict__ b2n,
    const float* __restrict__ m2, const float* __restrict__ v2,
    const float* __restrict__ Wc, const float* __restrict__ bc,
    const float* __restrict__ g3, const float* __restrict__ b3n,
    const float* __restrict__ m3, const float* __restrict__ v3,
    const float* __restrict__ Wf,
    f16* __restrict__ A1, f16* __restrict__ A2, f16* __restrict__ A3,
    f16* __restrict__ Wf16,
    float* __restrict__ beta1, float* __restrict__ beta2, float* __restrict__ beta3)
{
  const int tid = blockIdx.x * blockDim.x + threadIdx.x;
  const int nth = gridDim.x * blockDim.x;

  for (int o = tid; o < 36; o += nth) {
    beta1[o] = (bt[o] - m1[o]) * (g1[o] * rsqrtf(v1[o] + BN_EPS)) + b1n[o];
    beta2[o] = (bs[o] - m2[o]) * (g2[o] * rsqrtf(v2[o] + BN_EPS)) + b2n[o];
    beta3[o] = (bc[o] - m3[o]) * (g3[o] * rsqrtf(v3[o] + BN_EPS)) + b3n[o];
  }
  // A1: 3 mtiles x 3 ktiles, M=36, K=65 (zero-padded to 48x96)
  for (int idx = tid; idx < 9 * 512; idx += nth) {
    int fi = idx >> 9, lane = (idx >> 4) & 31, e = idx & 15;
    int mt = fi / 3, kt = fi % 3;
    int row = mt * 16 + (lane & 15);
    int k = kt * 32 + a_k(lane, e);
    float v = 0.f;
    if (row < 36 && k < 65) v = Wt[row * 65 + k] * (g1[row] * rsqrtf(v1[row] + BN_EPS));
    A1[idx] = (f16)v;
  }
  // A2: 3 x 27, M=36, K=864 (= i*24+h, exactly Ws row layout)
  for (int idx = tid; idx < 81 * 512; idx += nth) {
    int fi = idx >> 9, lane = (idx >> 4) & 31, e = idx & 15;
    int mt = fi / 27, kt = fi % 27;
    int row = mt * 16 + (lane & 15);
    int k = kt * 32 + a_k(lane, e);
    float v = 0.f;
    if (row < 36) v = Ws[row * 864 + k] * (g2[row] * rsqrtf(v2[row] + BN_EPS));
    A2[idx] = (f16)v;
  }
  // A3: 3 x 17, M=36, K=540 (= i*15+t, exactly Wc row layout)
  for (int idx = tid; idx < 51 * 512; idx += nth) {
    int fi = idx >> 9, lane = (idx >> 4) & 31, e = idx & 15;
    int mt = fi / 17, kt = fi % 17;
    int row = mt * 16 + (lane & 15);
    int k = kt * 32 + a_k(lane, e);
    float v = 0.f;
    if (row < 36 && k < 540) v = Wc[row * 540 + k] * (g3[row] * rsqrtf(v3[row] + BN_EPS));
    A3[idx] = (f16)v;
  }
  for (int idx = tid; idx < 288 * 288; idx += nth) Wf16[idx] = (f16)Wf[idx];
}

// ---------------------------------------------------------------------------
// Fused conv1(+BN+ELU) -> conv2(+BN+ELU) -> avgpool3, per (batch, 48-col chunk)
// conv1 GEMM: M=36(48) K=65(96) N=24*48 ; conv2 GEMM: M=36(48) K=864 N=48
// LDS: x window (f16) + conv1 activations (f16, [w][i*24+h]) + conv2 acts (f32)
// ---------------------------------------------------------------------------
__global__ void __launch_bounds__(384) conv12_kernel(
    const float* __restrict__ x, const f16* __restrict__ A1, const f16* __restrict__ A2,
    const float* __restrict__ beta1, const float* __restrict__ beta2,
    float* __restrict__ h2)
{
  extern __shared__ char smem[];
  f16*   xlds  = (f16*)smem;                      // 24 x 112 halves  (5376 B)
  f16*   h1lds = (f16*)(smem + 5376);             // 48 x 864 halves  (82944 B)
  float* act2  = (float*)(smem + 5376 + 82944);   // 36 x 48 floats   (6912 B)

  const int chunk = blockIdx.x;    // 0..9 (48 conv positions each)
  const int b     = blockIdx.y;    // 0..255
  const int tid   = threadIdx.x;
  const int wave  = tid >> 5;      // 12 waves
  const int lane  = tid & 31;
  const int lc    = lane & 15;
  const int lh    = lane >> 4;

  // Stage input window slice (x[..., 501:1001]) as f16, zero-pad past width 500
  for (int idx = tid; idx < 24 * 112; idx += 384) {
    int h = idx / 112, wl = idx % 112;
    int w = chunk * 48 + wl;
    float v = 0.f;
    if (w < 500) v = x[((size_t)b * 24 + h) * 1001 + 501 + w];
    xlds[idx] = (f16)v;
  }
  __syncthreads();

  // conv1: each wave owns n-tiles nt, nt+12, ... of 72 (N = 24 rows * 48 cols)
  v16h a1[3][3];
#pragma unroll
  for (int mt = 0; mt < 3; ++mt)
#pragma unroll
    for (int kt = 0; kt < 3; ++kt) a1[mt][kt] = ldfragA(A1, mt * 3 + kt, lane);

  for (int nt = wave; nt < 72; nt += 12) {
    int col = nt * 16 + lc;
    int hh = col / 48;
    int wl = col % 48;
    v16h bfr[3];
#pragma unroll
    for (int kt = 0; kt < 3; ++kt) {
      v16h bb;
#pragma unroll
      for (int e = 0; e < 16; ++e) {
        int k = kt * 32 + lh * 16 + e;   // B layout: N=lane%16, K=16*(lane/16)+e
        bb[e] = (k < 65) ? xlds[hh * 112 + wl + k] : (f16)0.f;
      }
      bfr[kt] = bb;
    }
#pragma unroll
    for (int mt = 0; mt < 3; ++mt) {
      v8f c = {};
#pragma unroll
      for (int kt = 0; kt < 3; ++kt) c = wmma_f16(a1[mt][kt], bfr[kt], c);
#pragma unroll
      for (int r = 0; r < 8; ++r) {         // C layout: M=r+8*(lane/16), N=lane%16
        int ch = mt * 16 + lh * 8 + r;
        if (ch < 36) h1lds[wl * 864 + ch * 24 + hh] = (f16)elu(c[r] + beta1[ch]);
      }
    }
  }
  __syncthreads();

  // conv2: 9 wave-jobs = 3 mtiles x 3 ntiles over N=48; B = aligned LDS vectors
  if (wave < 9) {
    int mt = wave / 3, nt2 = wave % 3;
    int wl = nt2 * 16 + lc;
    v8f c = {};
    for (int kt = 0; kt < 27; ++kt) {
      v16h bb = *(const v16h*)(h1lds + wl * 864 + kt * 32 + lh * 16);
      c = wmma_f16(ldfragA(A2, mt * 27 + kt, lane), bb, c);
    }
#pragma unroll
    for (int r = 0; r < 8; ++r) {
      int ch = mt * 16 + lh * 8 + r;
      if (ch < 36) act2[ch * 48 + wl] = elu(c[r] + beta2[ch]);
    }
  }
  __syncthreads();

  // avgpool width/3 -> h2[b][ch][j], j < 145 (stride-160 rows)
  for (int idx = tid; idx < 36 * 16; idx += 384) {
    int ch = idx / 16, jl = idx % 16;
    int j = chunk * 16 + jl;
    if (j < 145) {
      float s = act2[ch * 48 + 3 * jl] + act2[ch * 48 + 3 * jl + 1] + act2[ch * 48 + 3 * jl + 2];
      h2[((size_t)b * 36 + ch) * 160 + j] = s * (1.f / 3.f);
    }
  }
}

// ---------------------------------------------------------------------------
// conv3(+BN+ELU) + avgpool15 -> f (256x288). One block per batch.
// GEMM: M=36(48), K=540(544), N=131(144); 9 waves = 9 n-tiles.
// ---------------------------------------------------------------------------
__global__ void __launch_bounds__(288) conv3_kernel(
    const float* __restrict__ h2, const f16* __restrict__ A3,
    const float* __restrict__ beta3, float* __restrict__ fbuf)
{
  extern __shared__ char smem[];
  float* act3 = (float*)smem;            // 36 x 144 floats (20736 B)
  const int b    = blockIdx.x;
  const int tid  = threadIdx.x;
  const int wave = tid >> 5;             // n-tile 0..8
  const int lane = tid & 31;
  const int lc   = lane & 15, lh = lane >> 4;
  const int col  = wave * 16 + lc;       // output position j' (valid < 131)
  const float* h2b = h2 + (size_t)b * 36 * 160;

  for (int mt = 0; mt < 3; ++mt) {
    v8f c = {};
    for (int kt = 0; kt < 17; ++kt) {
      v16h bb;
#pragma unroll
      for (int e = 0; e < 16; ++e) {
        int k = kt * 32 + lh * 16 + e;   // k = i*15 + t
        float v = 0.f;
        if (k < 540 && col < 131) v = h2b[(k / 15) * 160 + col + (k % 15)];
        bb[e] = (f16)v;
      }
      c = wmma_f16(ldfragA(A3, mt * 17 + kt, lane), bb, c);
    }
#pragma unroll
    for (int r = 0; r < 8; ++r) {
      int ch = mt * 16 + lh * 8 + r;
      if (ch < 36) act3[ch * 144 + col] = elu(c[r] + beta3[ch]);
    }
  }
  __syncthreads();

  // avgpool/15 -> f[b][ch*8+s]  (reshape of (36,1,8))
  if (tid < 288) {
    int ch = tid / 8, s = tid % 8;
    float sum = 0.f;
#pragma unroll
    for (int q = 0; q < 15; ++q) sum += act3[ch * 144 + s * 15 + q];
    fbuf[(size_t)b * 288 + tid] = sum * (1.f / 15.f);
  }
}

// ---------------------------------------------------------------------------
// FC: out = (f @ Wf + bf + f) / 501.  M=256(16 mt) N=288(18 nt) K=288(9 kt)
// ---------------------------------------------------------------------------
__global__ void __launch_bounds__(256) fc_kernel(
    const float* __restrict__ fbuf, const f16* __restrict__ Wf16,
    const float* __restrict__ bf, float* __restrict__ out)
{
  const int tid  = threadIdx.x;
  const int lane = tid & 31;
  const int lc   = lane & 15, lh = lane >> 4;
  const int job  = blockIdx.x * 8 + (tid >> 5);   // 0..287
  const int mt   = job / 18, nt = job % 18;
  const int arow = mt * 16 + lc;

  v8f c = {};
  for (int kt = 0; kt < 9; ++kt) {
    v16h a, bb;
#pragma unroll
    for (int e = 0; e < 16; ++e) {
      int ka = kt * 32 + a_k(lane, e);
      a[e] = (f16)fbuf[(size_t)arow * 288 + ka];
      int kb = kt * 32 + lh * 16 + e;
      bb[e] = Wf16[(size_t)kb * 288 + nt * 16 + lc];
    }
    c = wmma_f16(a, bb, c);
  }
#pragma unroll
  for (int r = 0; r < 8; ++r) {
    int brow = mt * 16 + lh * 8 + r;
    int col  = nt * 16 + lc;
    float pred = c[r] + bf[col];
    out[(size_t)brow * 288 + col] = (pred + fbuf[(size_t)brow * 288 + col]) * (1.f / 501.f);
  }
}

// ---------------------------------------------------------------------------
extern "C" void kernel_launch(void* const* d_in, const int* in_sizes, int n_in,
                              void* d_out, int out_size, void* d_ws, size_t ws_size,
                              hipStream_t stream) {
  const float* x   = (const float*)d_in[0];
  const float* Wt  = (const float*)d_in[1];
  const float* bt  = (const float*)d_in[2];
  const float* g1  = (const float*)d_in[3];
  const float* b1  = (const float*)d_in[4];
  const float* m1  = (const float*)d_in[5];
  const float* v1  = (const float*)d_in[6];
  const float* Ws_ = (const float*)d_in[7];
  const float* bs  = (const float*)d_in[8];
  const float* g2  = (const float*)d_in[9];
  const float* b2  = (const float*)d_in[10];
  const float* m2  = (const float*)d_in[11];
  const float* v2  = (const float*)d_in[12];
  const float* Wc  = (const float*)d_in[13];
  const float* bc  = (const float*)d_in[14];
  const float* g3  = (const float*)d_in[15];
  const float* b3  = (const float*)d_in[16];
  const float* m3  = (const float*)d_in[17];
  const float* v3  = (const float*)d_in[18];
  const float* Wf  = (const float*)d_in[19];
  const float* bf  = (const float*)d_in[20];
  float* out = (float*)d_out;
  (void)in_sizes; (void)n_in; (void)out_size; (void)ws_size;

  char* ws = (char*)d_ws;
  size_t off = 0;
  auto carve = [&](size_t bytes) -> char* {
    char* p = ws + off;
    off = (off + bytes + 255) & ~(size_t)255;
    return p;
  };
  f16*   A1    = (f16*)carve((size_t)9 * 512 * 2);
  f16*   A2    = (f16*)carve((size_t)81 * 512 * 2);
  f16*   A3    = (f16*)carve((size_t)51 * 512 * 2);
  f16*   Wf16  = (f16*)carve((size_t)288 * 288 * 2);
  float* beta1 = (float*)carve(36 * 4);
  float* beta2 = (float*)carve(36 * 4);
  float* beta3 = (float*)carve(36 * 4);
  float* h2    = (float*)carve((size_t)256 * 36 * 160 * 4);  // ~5.9 MB
  float* fbuf  = (float*)carve((size_t)256 * 288 * 4);

  prep_kernel<<<64, 256, 0, stream>>>(Wt, bt, g1, b1, m1, v1,
                                      Ws_, bs, g2, b2, m2, v2,
                                      Wc, bc, g3, b3, m3, v3, Wf,
                                      A1, A2, A3, Wf16, beta1, beta2, beta3);

  size_t smem12 = 24 * 112 * 2 + 48 * 864 * 2 + 36 * 48 * 4;  // 95232 B
  conv12_kernel<<<dim3(10, 256), 384, smem12, stream>>>(x, A1, A2, beta1, beta2, h2);

  conv3_kernel<<<256, 288, 36 * 144 * 4, stream>>>(h2, A3, beta3, fbuf);

  fc_kernel<<<36, 256, 0, stream>>>(fbuf, Wf16, bf, out);
}
